// FBSDEModel_MFG_86062554677933
// MI455X (gfx1250) — matrix operand order repaired and checked by reading, not compiled, not run on hardware
//
#include <hip/hip_runtime.h>
#include <hip/hip_bf16.h>

typedef __attribute__((ext_vector_type(16))) _Float16 v16h;
typedef __attribute__((ext_vector_type(8)))  _Float16 v8h;
typedef __attribute__((ext_vector_type(8)))  float    v8f;

#define N_SAMPLE 16384
#define TSTEPS   64
#define HID      128
#define CA_      1.0f
#define CX_      0.1f
#define GAMMA_   0.5f
#define SIGMA_   0.7f
#define CG_      1.0f

// workspace byte offsets
#define WS_X    0
#define WS_Y    (N_SAMPLE * 4)
#define WS_SUM  (2 * N_SAMPLE * 4)
#define WS_FRAG (2 * N_SAMPLE * 4 + 1024)

// ---------------------------------------------------------------------------
// prep: zero per-step sum slots; swizzle z-net W1 (f32 row-major [k][n]) into
// f16 WMMA B-fragments. Fragment f = nt*4+kc covers N-tile nt (16 cols) and
// K-chunk kc (32 rows). Lane l<16: N = nt*16+l, K = kc*32 + 0..15;
// lane l>=16: N = nt*16+(l-16), K = kc*32 + 16..31. 16 halfs per lane.
// ---------------------------------------------------------------------------
__global__ void __launch_bounds__(256)
prep_kernel(const float* __restrict__ z_w1, _Float16* __restrict__ frag,
            float* __restrict__ sums) {
    int tid = blockIdx.x * blockDim.x + threadIdx.x;
    if (blockIdx.x == 0 && threadIdx.x <= TSTEPS) sums[threadIdx.x] = 0.0f;
    if (tid < HID * HID) {
        int f = tid >> 9;          // fragment index (0..31)
        int l = (tid >> 4) & 31;   // lane
        int e = tid & 15;          // element within v16h
        int nt = f >> 2, kc = f & 3;
        int n = nt * 16 + (l & 15);
        int k = kc * 32 + ((l & 16) ? 16 : 0) + e;
        frag[tid] = (_Float16)z_w1[k * HID + n];
    }
}

// ---------------------------------------------------------------------------
// init: y0 = yMLP(x) (1 -> 64 -> 64 -> 1), copy x into workspace, seed sum[0].
// One-time cost; plain VALU with weights staged in LDS.
// ---------------------------------------------------------------------------
__global__ void __launch_bounds__(256)
init_y_kernel(const float* __restrict__ x_in,
              const float* __restrict__ w0, const float* __restrict__ b0,
              const float* __restrict__ w1, const float* __restrict__ b1,
              const float* __restrict__ w2, const float* __restrict__ b2p,
              float* __restrict__ ws_x, float* __restrict__ ws_y,
              float* __restrict__ sums) {
    __shared__ float sw1[64 * 64];
    __shared__ float sw0[64], sb0[64], sb1[64], sw2[64];
    int tid = threadIdx.x;
    for (int i = tid; i < 64 * 64; i += 256) sw1[i] = w1[i];
    if (tid < 64) { sw0[tid] = w0[tid]; sb0[tid] = b0[tid];
                    sb1[tid] = b1[tid]; sw2[tid] = w2[tid]; }
    __syncthreads();

    int g = blockIdx.x * 256 + tid;
    float xv = x_in[g];
    float h1[64];
#pragma unroll
    for (int j = 0; j < 64; ++j)
        h1[j] = fmaxf(fmaf(xv, sw0[j], sb0[j]), 0.0f);

    float y = b2p[0];
    for (int j2 = 0; j2 < 64; ++j2) {
        float acc = sb1[j2];
#pragma unroll 8
        for (int j = 0; j < 64; ++j)
            acc = fmaf(h1[j], sw1[j * 64 + j2], acc);
        y = fmaf(fmaxf(acc, 0.0f), sw2[j2], y);
    }
    ws_x[g] = xv;
    ws_y[g] = y;

    float s = y;
#pragma unroll
    for (int off = 16; off >= 1; off >>= 1) s += __shfl_xor(s, off, 32);
    if ((tid & 31) == 0) atomicAdd(&sums[0], s);
}

// ---------------------------------------------------------------------------
// step s: zin = [t, x]; h1 = relu(x*W0[1]+t*W0[0]+b0) in LDS (f16, padded);
// h2 GEMM via v_wmma_f32_16x16x32_f16 (8 N-tiles x 4 K-chunks per wave);
// y_diff = relu(h2+b1) . w2 + b2 folded into epilogue + shfl_xor reduce;
// Euler update of (x, y); batch-sum of y into sums[s+1] via f32 atomics.
// ---------------------------------------------------------------------------
__global__ void __launch_bounds__(256)
step_kernel(const float* __restrict__ dw,
            const float* __restrict__ z_w0, const float* __restrict__ z_b0,
            const float* __restrict__ z_b1, const float* __restrict__ z_w2,
            const float* __restrict__ z_b2,
            const _Float16* __restrict__ frag,
            float* __restrict__ ws_x, float* __restrict__ ws_y,
            float* __restrict__ sums, float* __restrict__ out, int s) {
    // 128 rows x (128+8) halfs: row stride 272B => lane bank stride 4, conflict-free b128
    __shared__ _Float16 h1s[128 * 136];
    __shared__ float cS[128], w0xS[128], b1S[128], w2S[128], ydiffS[128];

    const float dt = 1.0f / (float)TSTEPS;
    const float t  = (float)s * dt;
    int tid   = threadIdx.x;
    int gbase = blockIdx.x * 128;

    if (tid < 128) {
        cS[tid]   = fmaf(t, z_w0[tid], z_b0[tid]);  // t*W0[0,j] + b0[j]
        w0xS[tid] = z_w0[128 + tid];                // W0[1,j]
        b1S[tid]  = z_b1[tid];
        w2S[tid]  = z_w2[tid];
    }
    __syncthreads();

    { // layer 1: each thread computes 64 features of one sample
        int i  = tid >> 1;
        int j0 = (tid & 1) * 64;
        float xv = ws_x[gbase + i];
#pragma unroll 16
        for (int j = 0; j < 64; ++j) {
            float h = fmaxf(fmaf(xv, w0xS[j0 + j], cS[j0 + j]), 0.0f);
            h1s[i * 136 + j0 + j] = (_Float16)h;
        }
    }
    __syncthreads();

    // layer 2 (WMMA) + layer 3 epilogue
    int wave = tid >> 5;
    int lane = tid & 31;
    int row  = wave * 16 + (lane & 15);
    int csel = (lane & 16) ? 16 : 0;  // A layout: hi half-wave holds K+16..K+31

    v16h A[4];
#pragma unroll
    for (int kc = 0; kc < 4; ++kc) {
        const _Float16* base = &h1s[row * 136 + kc * 32 + csel];
        v8h lo = *(const v8h*)(base);
        v8h hi = *(const v8h*)(base + 8);
        v16h a;
#pragma unroll
        for (int q = 0; q < 8; ++q) { a[q] = lo[q]; a[q + 8] = hi[q]; }
        A[kc] = a;
    }

    float part[8];
#pragma unroll
    for (int r = 0; r < 8; ++r) part[r] = 0.0f;

    const v16h* B = (const v16h*)frag;
#pragma unroll
    for (int nt = 0; nt < 8; ++nt) {
        v8f acc = {};
#pragma unroll
        for (int kc = 0; kc < 4; ++kc) {
            v16h b = B[(nt * 4 + kc) * 32 + lane];
            acc = __builtin_amdgcn_wmma_f32_16x16x32_f16(
                false, A[kc], false, b, (short)0, acc, false, false);
        }
        int n = nt * 16 + (lane & 15);
        float bias = b1S[n], w2v = w2S[n];
#pragma unroll
        for (int r = 0; r < 8; ++r) {
            float h2 = fmaxf(acc[r] + bias, 0.0f);
            part[r] = fmaf(h2, w2v, part[r]);
        }
    }
    // reduce over the 16 lanes of each half-wave (cols); rows 0-7 live in
    // lanes 0-15, rows 8-15 in lanes 16-31 (f32 C/D layout)
#pragma unroll
    for (int off = 8; off >= 1; off >>= 1)
#pragma unroll
        for (int r = 0; r < 8; ++r)
            part[r] += __shfl_xor(part[r], off, 32);

    float b2v = z_b2[0];
    if (lane == 0) {
#pragma unroll
        for (int r = 0; r < 8; ++r) ydiffS[wave * 16 + r] = part[r] + b2v;
    }
    if (lane == 16) {
#pragma unroll
        for (int r = 0; r < 8; ++r) ydiffS[wave * 16 + 8 + r] = part[r] + b2v;
    }
    __syncthreads();

    // Euler update
    float ysum = 0.0f;
    if (tid < 128) {
        int g = gbase + tid;
        float xv  = ws_x[g];
        float yv  = ws_y[g];
        float dwv = dw[g * TSTEPS + s];
        float abar   = -sums[s] * (1.0f / (float)N_SAMPLE);  // mean(-y/ca)
        float a      = -yv / CA_;
        float ydrift = -(CX_ * xv - GAMMA_ * abar);
        float xn = xv + a * dt + SIGMA_ * dwv;
        float yn = yv + ydrift * dt + ydiffS[tid] * dwv;
        ws_x[g] = xn;
        ws_y[g] = yn;
        if (s == TSTEPS - 1) {
            out[g]            = yn;        // y_T
            out[N_SAMPLE + g] = CG_ * xn;  // terminal_cost_dx(x_T)
        }
        ysum = yn;
    }
#pragma unroll
    for (int off = 16; off >= 1; off >>= 1) ysum += __shfl_xor(ysum, off, 32);
    if ((tid & 31) == 0 && tid < 128) atomicAdd(&sums[s + 1], ysum);
}

// ---------------------------------------------------------------------------
extern "C" void kernel_launch(void* const* d_in, const int* in_sizes, int n_in,
                              void* d_out, int out_size, void* d_ws, size_t ws_size,
                              hipStream_t stream) {
    const float* x_in = (const float*)d_in[0];
    const float* dw   = (const float*)d_in[1];
    const float* y_w0 = (const float*)d_in[2];
    const float* y_b0 = (const float*)d_in[3];
    const float* y_w1 = (const float*)d_in[4];
    const float* y_b1 = (const float*)d_in[5];
    const float* y_w2 = (const float*)d_in[6];
    const float* y_b2 = (const float*)d_in[7];
    const float* z_w0 = (const float*)d_in[8];
    const float* z_b0 = (const float*)d_in[9];
    const float* z_w1 = (const float*)d_in[10];
    const float* z_b1 = (const float*)d_in[11];
    const float* z_w2 = (const float*)d_in[12];
    const float* z_b2 = (const float*)d_in[13];

    char* ws = (char*)d_ws;
    float*    ws_x = (float*)(ws + WS_X);
    float*    ws_y = (float*)(ws + WS_Y);
    float*    sums = (float*)(ws + WS_SUM);
    _Float16* frag = (_Float16*)(ws + WS_FRAG);
    float*    out  = (float*)d_out;

    prep_kernel<<<64, 256, 0, stream>>>(z_w1, frag, sums);
    init_y_kernel<<<N_SAMPLE / 256, 256, 0, stream>>>(
        x_in, y_w0, y_b0, y_w1, y_b1, y_w2, y_b2, ws_x, ws_y, sums);
    for (int s = 0; s < TSTEPS; ++s) {
        step_kernel<<<N_SAMPLE / 128, 256, 0, stream>>>(
            dw, z_w0, z_b0, z_b1, z_w2, z_b2, frag, ws_x, ws_y, sums, out, s);
    }
}
